// Decoder_Block_22179211116672
// MI455X (gfx1250) — compile-verified, matrix operand order checked by reading, not run on hardware
//
#include <hip/hip_runtime.h>
#include <math.h>

typedef __attribute__((ext_vector_type(16))) _Float16 v16h;
typedef __attribute__((ext_vector_type(8)))  _Float16 v8h;
typedef __attribute__((ext_vector_type(8)))  float    v8f;

// Model constants
#define NB     8
#define N_IN   1024
#define N_OUT  4096
#define D_IN   512
#define D_OUT  256

// LDS tile row stride in halves: 32 K-halves + 8 pad, keeps 16B alignment (80B rows)
#define LDK 40

// Build a 16-half WMMA fragment from LDS: two aligned 16-byte loads (ds_load_b128).
__device__ __forceinline__ v16h frag_lds(const _Float16* base)
{
    v8h lo = *(const v8h*)(base);
    v8h hi = *(const v8h*)(base + 16);
    return __builtin_shufflevector(lo, hi, 0, 1, 2, 3, 4, 5, 6, 7,
                                   8, 9, 10, 11, 12, 13, 14, 15);
}

// Pack 8 floats into 8 halves (lowers to v_cvt_pk_f16_f32 x4).
__device__ __forceinline__ v8h cvt8(float4 a, float4 b)
{
    v8h h;
    h[0] = (_Float16)a.x; h[1] = (_Float16)a.y; h[2] = (_Float16)a.z; h[3] = (_Float16)a.w;
    h[4] = (_Float16)b.x; h[5] = (_Float16)b.y; h[6] = (_Float16)b.z; h[7] = (_Float16)b.w;
    return h;
}

// ---------------------------------------------------------------------------
// Batched WMMA GEMM:  C[m,n] = sum_k opA(m,k) * opB(n,k) + bias[n] (+ res)
//   opA(m,k) = TA ? A[k*lda+m] : A[m*lda+k]
//   opB(n,k) = TB ? B[k*ldb+n] : B[n*ldb+k]   (default: weight stored [N,K])
// Block: 256 threads (8 waves) -> 64x64 C tile; K-step 32 staged in LDS as f16.
// Wave w: row-tile (w&3)*16, col-tiles (w>>2)*32 + {0,16}; shared A fragment,
// two v_wmma_f32_16x16x32_f16 per K step.
// REQUIRES (guaranteed by this model): K % 32 == 0, M % 16 == 0, N % 16 == 0,
// lda/ldb % 4 == 0, 16B-aligned base pointers.
// ---------------------------------------------------------------------------
template <int TA, int TB>
__global__ __launch_bounds__(256) void k_gemm_wmma(
    const float* __restrict__ A, long lda, long strideA,
    const float* __restrict__ Bm, long ldb, long strideB,
    const float* __restrict__ bias,
    const float* __restrict__ res, long ldres, long strideRes,
    float* __restrict__ C, long ldc, long strideC,
    int M, int N, int K)
{
    __shared__ _Float16 Alds[64 * LDK];
    __shared__ _Float16 Blds[64 * LDK];

    const int bz = blockIdx.z;
    A  += (long)bz * strideA;
    Bm += (long)bz * strideB;
    C  += (long)bz * strideC;
    if (res) res += (long)bz * strideRes;

    const int m0  = blockIdx.y * 64;
    const int n0  = blockIdx.x * 64;
    const int tid = threadIdx.x;

    const int lane = tid & 31;
    const int w    = tid >> 5;
    const int r    = lane & 15;
    const int g    = lane >> 4;
    const int tm   = (w & 3) * 16;      // wave's row-tile offset within block
    const int tn   = (w >> 2) * 32;     // wave's col offset (two 16-wide tiles)

    // Staging coordinates (hoisted out of the K loop).
    const int srow = tid >> 2;          // 0..63   (row-major staging)
    const int skb  = (tid & 3) * 8;     // 0,8,16,24
    const int skk  = tid >> 3;          // 0..31   (transposed staging)
    const int srb  = (tid & 7) * 8;     // 0..56

    v8f acc0 = {0.f, 0.f, 0.f, 0.f, 0.f, 0.f, 0.f, 0.f};
    v8f acc1 = {0.f, 0.f, 0.f, 0.f, 0.f, 0.f, 0.f, 0.f};

    for (int k0 = 0; k0 < K; k0 += 32) {
        // ---- stage A[64x32] tile (f32 -> f16), branch hoisted, b128 traffic
        if (TA == 0) {
            float4 a0 = {0.f, 0.f, 0.f, 0.f}, a1 = {0.f, 0.f, 0.f, 0.f};
            if (m0 + srow < M) {
                const float4* src = (const float4*)(A + (long)(m0 + srow) * lda + k0 + skb);
                a0 = src[0]; a1 = src[1];
            }
            *(v8h*)&Alds[srow * LDK + skb] = cvt8(a0, a1);       // ds_store_b128
        } else {
            float4 a0 = {0.f, 0.f, 0.f, 0.f}, a1 = {0.f, 0.f, 0.f, 0.f};
            if (m0 + srb < M) {   // uniform for the whole group of 8 (M%16==0)
                const float4* src = (const float4*)(A + (long)(k0 + skk) * lda + m0 + srb);
                a0 = src[0]; a1 = src[1];
            }
            const v8h h = cvt8(a0, a1);
#pragma unroll
            for (int i = 0; i < 8; ++i) Alds[(srb + i) * LDK + skk] = h[i];
        }
        // ---- stage B[64x32] tile
        if (TB == 0) {
            float4 b0 = {0.f, 0.f, 0.f, 0.f}, b1 = {0.f, 0.f, 0.f, 0.f};
            if (n0 + srow < N) {
                const float4* src = (const float4*)(Bm + (long)(n0 + srow) * ldb + k0 + skb);
                b0 = src[0]; b1 = src[1];
            }
            *(v8h*)&Blds[srow * LDK + skb] = cvt8(b0, b1);
        } else {
            float4 b0 = {0.f, 0.f, 0.f, 0.f}, b1 = {0.f, 0.f, 0.f, 0.f};
            if (n0 + srb < N) {
                const float4* src = (const float4*)(Bm + (long)(k0 + skk) * ldb + n0 + srb);
                b0 = src[0]; b1 = src[1];
            }
            const v8h h = cvt8(b0, b1);
#pragma unroll
            for (int i = 0; i < 8; ++i) Blds[(srb + i) * LDK + skk] = h[i];
        }
        __syncthreads();

        // ---- fragments + 2 WMMAs (shared A fragment)
        // A-frag layout (CDNA5 ISA): lane r row, lanes 0-15 K0..7/16..23,
        // lanes 16-31 K8..15/24..31 -> LDS offsets 8*g and 8*g+16.
        const v16h af  = frag_lds(&Alds[(tm + r) * LDK + 8 * g]);
        const v16h bf0 = frag_lds(&Blds[(tn + r) * LDK + 8 * g]);
        const v16h bf1 = frag_lds(&Blds[(tn + 16 + r) * LDK + 8 * g]);
        acc0 = __builtin_amdgcn_wmma_f32_16x16x32_f16(false, af, false, bf0,
                                                      (short)0, acc0, false, false);
        acc1 = __builtin_amdgcn_wmma_f32_16x16x32_f16(false, af, false, bf1,
                                                      (short)0, acc1, false, false);
        __syncthreads();
    }

    // ---- epilogue: bias + residual + store (per-16-tile bounds only)
#pragma unroll
    for (int i = 0; i < 8; ++i) {
        const int m = m0 + tm + i + 8 * g;   // C/D layout: VGPR i -> M=i (+8 hi half)
        if (m < M) {
            const int n_a = n0 + tn + r;
            const int n_b = n_a + 16;
            if (n_a < N) {
                float y = acc0[i];
                if (bias) y += bias[n_a];
                if (res)  y += res[(long)m * ldres + n_a];
                C[(long)m * ldc + n_a] = y;
            }
            if (n_b < N) {
                float y = acc1[i];
                if (bias) y += bias[n_b];
                if (res)  y += res[(long)m * ldres + n_b];
                C[(long)m * ldc + n_b] = y;
            }
        }
    }
}

// ---------------------------------------------------------------------------
// LayerNorm over last dim; one block (256 thr) per row. In-place safe.
// ---------------------------------------------------------------------------
__global__ __launch_bounds__(256) void k_ln(
    const float* __restrict__ x, const float* __restrict__ w,
    const float* __restrict__ b, float* __restrict__ y, int C)
{
    const long row = blockIdx.x;
    const float* xr = x + row * (long)C;
    float* yr = y + row * (long)C;
    __shared__ float red[256];
    const int tid = threadIdx.x;

    float s = 0.f;
    for (int c = tid; c < C; c += 256) s += xr[c];
    red[tid] = s; __syncthreads();
    for (int o = 128; o > 0; o >>= 1) { if (tid < o) red[tid] += red[tid + o]; __syncthreads(); }
    const float mean = red[0] / (float)C; __syncthreads();

    float v = 0.f;
    for (int c = tid; c < C; c += 256) { float d = xr[c] - mean; v += d * d; }
    red[tid] = v; __syncthreads();
    for (int o = 128; o > 0; o >>= 1) { if (tid < o) red[tid] += red[tid + o]; __syncthreads(); }
    const float rstd = rsqrtf(red[0] / (float)C + 1e-5f);

    for (int c = tid; c < C; c += 256) yr[c] = (xr[c] - mean) * rstd * w[c] + b[c];
}

// ---------------------------------------------------------------------------
// Softmax over last dim; optional per-row-group scale (channel-attn temp).
// ---------------------------------------------------------------------------
__global__ __launch_bounds__(256) void k_softmax_last(
    const float* __restrict__ x, float* __restrict__ y, int C,
    const float* __restrict__ scale, int scaleDiv, int scalePeriod)
{
    const long row = blockIdx.x;
    float sc = 1.f;
    if (scale) sc = scale[(int)((row / scaleDiv) % scalePeriod)];
    const float* xr = x + row * (long)C;
    float* yr = y + row * (long)C;
    __shared__ float red[256];
    const int tid = threadIdx.x;

    float mx = -3.4e38f;
    for (int c = tid; c < C; c += 256) mx = fmaxf(mx, xr[c] * sc);
    red[tid] = mx; __syncthreads();
    for (int o = 128; o > 0; o >>= 1) { if (tid < o) red[tid] = fmaxf(red[tid], red[tid + o]); __syncthreads(); }
    const float rmax = red[0]; __syncthreads();

    float ss = 0.f;
    for (int c = tid; c < C; c += 256) ss += expf(xr[c] * sc - rmax);
    red[tid] = ss; __syncthreads();
    for (int o = 128; o > 0; o >>= 1) { if (tid < o) red[tid] += red[tid + o]; __syncthreads(); }
    const float inv = 1.f / red[0];

    for (int c = tid; c < C; c += 256) yr[c] = expf(xr[c] * sc - rmax) * inv;
}

// Softmax over token axis (axis=1) of [B,N,C]; one thread per (b,c) column.
__global__ void k_softmax_mid(const float* __restrict__ x, float* __restrict__ y,
                              int Bn, int N, int C)
{
    const long idx = blockIdx.x * (long)blockDim.x + threadIdx.x;
    if (idx >= (long)Bn * C) return;
    const long bb = idx / C;
    const int  c  = (int)(idx % C);
    const float* xp = x + bb * (long)N * C + c;
    float* yp = y + bb * (long)N * C + c;
    float mx = -3.4e38f;
    for (int n = 0; n < N; ++n) mx = fmaxf(mx, xp[(long)n * C]);
    float s = 0.f;
    for (int n = 0; n < N; ++n) s += expf(xp[(long)n * C] - mx);
    const float inv = 1.f / s;
    for (int n = 0; n < N; ++n) yp[(long)n * C] = expf(xp[(long)n * C] - mx) * inv;
}

// Depthwise 3x3 conv SAME on [B, H*W, C] channel-last layout; w is [C,1,3,3].
__global__ void k_dwconv3(const float* __restrict__ x, const float* __restrict__ w,
                          const float* __restrict__ b, float* __restrict__ y,
                          int Bn, int H, int W, int C)
{
    const long idx = blockIdx.x * (long)blockDim.x + threadIdx.x;
    const long total = (long)Bn * H * W * C;
    if (idx >= total) return;
    const int  c  = (int)(idx % C);
    long rest = idx / C;
    const int  xw = (int)(rest % W); rest /= W;
    const int  yh = (int)(rest % H);
    const long bb = rest / H;
    float acc = b[c];
#pragma unroll
    for (int dy = -1; dy <= 1; ++dy) {
#pragma unroll
        for (int dx = -1; dx <= 1; ++dx) {
            const int yy = yh + dy, xx = xw + dx;
            if (yy >= 0 && yy < H && xx >= 0 && xx < W)
                acc += x[((bb * H * W) + (long)yy * W + xx) * C + c] * w[c * 9 + (dy + 1) * 3 + (dx + 1)];
        }
    }
    y[idx] = acc;
}

__global__ void k_add(const float* __restrict__ a, const float* __restrict__ b,
                      float* __restrict__ o, long n)
{
    const long i = blockIdx.x * (long)blockDim.x + threadIdx.x;
    if (i < n) o[i] = a[i] + b[i];
}

__global__ void k_gelu(float* __restrict__ x, long n)
{
    const long i = blockIdx.x * (long)blockDim.x + threadIdx.x;
    if (i < n) { float v = x[i]; x[i] = 0.5f * v * (1.f + erff(v * 0.70710678118654752f)); }
}

// atten = (a1 + a2) * sigmoid(w[0]) + x
__global__ void k_merge3(const float* __restrict__ a1, const float* __restrict__ a2,
                         const float* __restrict__ x, const float* __restrict__ ws,
                         float* __restrict__ o, long n)
{
    const long i = blockIdx.x * (long)blockDim.x + threadIdx.x;
    if (i < n) {
        const float f = 1.f / (1.f + expf(-ws[0]));
        o[i] = (a1[i] + a2[i]) * f + x[i];
    }
}

// qkv [B,N,3C] -> qn,kn,v in [B*h, hd, N] with L2 normalization over N for q,k.
__global__ __launch_bounds__(256) void k_headprep(
    const float* __restrict__ qkv, float* __restrict__ qn, float* __restrict__ kn,
    float* __restrict__ vh, int N, int C, int heads)
{
    const int hd = C / heads;
    const long bd = blockIdx.x;               // (b*heads + h)*hd + d
    const int d  = (int)(bd % hd);
    long rest = bd / hd;
    const int hh = (int)(rest % heads);
    const long bb = rest / heads;
    const long row3 = 3L * C;
    const float* base = qkv + ((long)bb * N) * row3 + (long)hh * hd + d;
    __shared__ float rq[256], rk[256];
    const int tid = threadIdx.x;
    float sq = 0.f, sk = 0.f;
    for (int n = tid; n < N; n += 256) {
        const float q = base[(long)n * row3];
        const float k = base[(long)n * row3 + C];
        sq += q * q; sk += k * k;
    }
    rq[tid] = sq; rk[tid] = sk; __syncthreads();
    for (int o = 128; o > 0; o >>= 1) {
        if (tid < o) { rq[tid] += rq[tid + o]; rk[tid] += rk[tid + o]; }
        __syncthreads();
    }
    const float qi = 1.f / fmaxf(sqrtf(rq[0]), 1e-12f);
    const float ki = 1.f / fmaxf(sqrtf(rk[0]), 1e-12f);
    const long orow = bd * (long)N;
    for (int n = tid; n < N; n += 256) {
        const long src = (long)n * row3;
        qn[orow + n] = base[src] * qi;
        kn[orow + n] = base[src + C] * ki;
        vh[orow + n] = base[src + 2 * C];
    }
}

// [B,h,hd,N] -> [B,N,C]
__global__ void k_untranspose(const float* __restrict__ oh, float* __restrict__ out,
                              int Bn, int N, int C, int heads)
{
    const long i = blockIdx.x * (long)blockDim.x + threadIdx.x;
    const long total = (long)Bn * N * C;
    if (i >= total) return;
    const int c = (int)(i % C);
    long rest = i / C;
    const int n = (int)(rest % N);
    const long bb = rest / N;
    const int hd = C / heads;
    const int hh = c / hd, d = c % hd;
    out[i] = oh[(((long)bb * heads + hh) * hd + d) * (long)N + n];
}

// ---------------------------------------------------------------------------
// Host-side parameter unpacking (jax pytree sorted-key flatten order)
// ---------------------------------------------------------------------------
struct MixP  { const float *dw_b,*dw_w,*fc1_b,*fc1_w,*fc2_b,*fc2_w,*ln_b,*ln_w; };
struct CaP   { const float *proj_b,*proj_w,*qkv_w,*temp; };
struct EaP   { const float *k_b,*k_w,*q_b,*q_w,*r_b,*r_w,*v_b,*v_w; };
struct DualP { CaP ca; EaP ea; MixP mlp1, mlp2;
               const float *n1_b,*n1_w,*n2_b,*n2_w,*n3_b,*n3_w,*n4_b,*n4_w; };
struct MabP  { const float *ca1_b,*ca1_w,*ca2_b,*ca2_w; MixP ffn;
               const float *lin_w,*norm_b,*norm_w,*pe_conv_w,*pe_lin_w,*pe_norm_b,*pe_norm_w,*w; };

struct Rd { void* const* d; int i; const float* next() { return (const float*)d[i++]; } };

static MixP rdMix(Rd& r) {
    MixP m; m.dw_b=r.next(); m.dw_w=r.next(); m.fc1_b=r.next(); m.fc1_w=r.next();
    m.fc2_b=r.next(); m.fc2_w=r.next(); m.ln_b=r.next(); m.ln_w=r.next(); return m;
}
static DualP rdDual(Rd& r) {
    DualP d;
    d.ca.proj_b=r.next(); d.ca.proj_w=r.next(); d.ca.qkv_w=r.next(); d.ca.temp=r.next();
    d.ea.k_b=r.next(); d.ea.k_w=r.next(); d.ea.q_b=r.next(); d.ea.q_w=r.next();
    d.ea.r_b=r.next(); d.ea.r_w=r.next(); d.ea.v_b=r.next(); d.ea.v_w=r.next();
    d.mlp1=rdMix(r); d.mlp2=rdMix(r);
    d.n1_b=r.next(); d.n1_w=r.next(); d.n2_b=r.next(); d.n2_w=r.next();
    d.n3_b=r.next(); d.n3_w=r.next(); d.n4_b=r.next(); d.n4_w=r.next();
    return d;
}
static MabP rdMab(Rd& r) {
    MabP m;
    m.ca1_b=r.next(); m.ca1_w=r.next(); m.ca2_b=r.next(); m.ca2_w=r.next();
    m.ffn=rdMix(r);
    m.lin_w=r.next(); m.norm_b=r.next(); m.norm_w=r.next();
    m.pe_conv_w=r.next(); m.pe_lin_w=r.next(); m.pe_norm_b=r.next(); m.pe_norm_w=r.next();
    m.w=r.next();
    return m;
}

// ---------------------------------------------------------------------------
// Launch helpers
// ---------------------------------------------------------------------------
static void gemm(hipStream_t s,
                 const float* A, long lda, long sA, int tA,
                 const float* Bm, long ldb, long sB, int tB,
                 const float* bias,
                 const float* res, long ldres, long sRes,
                 float* C, long ldc, long sC,
                 int M, int N, int K, int batches)
{
    dim3 grid((N + 63) / 64, (M + 63) / 64, batches);
    dim3 blk(256);
    if (tA == 0 && tB == 0)
        k_gemm_wmma<0,0><<<grid, blk, 0, s>>>(A, lda, sA, Bm, ldb, sB, bias, res, ldres, sRes, C, ldc, sC, M, N, K);
    else if (tA == 0 && tB == 1)
        k_gemm_wmma<0,1><<<grid, blk, 0, s>>>(A, lda, sA, Bm, ldb, sB, bias, res, ldres, sRes, C, ldc, sC, M, N, K);
    else if (tA == 1 && tB == 0)
        k_gemm_wmma<1,0><<<grid, blk, 0, s>>>(A, lda, sA, Bm, ldb, sB, bias, res, ldres, sRes, C, ldc, sC, M, N, K);
    else
        k_gemm_wmma<1,1><<<grid, blk, 0, s>>>(A, lda, sA, Bm, ldb, sB, bias, res, ldres, sRes, C, ldc, sC, M, N, K);
}
static void ln(hipStream_t s, const float* x, const float* w, const float* b,
               float* y, long rows, int C)
{ k_ln<<<dim3((unsigned)rows), 256, 0, s>>>(x, w, b, y, C); }

static void smax_last(hipStream_t s, const float* x, float* y, long rows, int C,
                      const float* scale, int sdiv, int sper)
{ k_softmax_last<<<dim3((unsigned)rows), 256, 0, s>>>(x, y, C, scale, sdiv, sper); }

static void smax_mid(hipStream_t s, const float* x, float* y, int Bn, int N, int C)
{ long t = (long)Bn * C; k_softmax_mid<<<dim3((unsigned)((t + 255) / 256)), 256, 0, s>>>(x, y, Bn, N, C); }

static void dwconv(hipStream_t s, const float* x, const float* w, const float* b,
                   float* y, int Bn, int H, int W, int C)
{ long t = (long)Bn * H * W * C; k_dwconv3<<<dim3((unsigned)((t + 255) / 256)), 256, 0, s>>>(x, w, b, y, Bn, H, W, C); }

static void addk(hipStream_t s, const float* a, const float* b, float* o, long n)
{ k_add<<<dim3((unsigned)((n + 255) / 256)), 256, 0, s>>>(a, b, o, n); }

static void geluk(hipStream_t s, float* x, long n)
{ k_gelu<<<dim3((unsigned)((n + 255) / 256)), 256, 0, s>>>(x, n); }

// mixffn: out = fc2( gelu( ln( dwconv(h) + h ) ) ) + resid, h = fc1(xn)
static void run_mixffn(hipStream_t s, const MixP& P, const float* xn, const float* resid,
                       float* out, float* Hh, float* Cc,
                       int Ntok, int Din, int Dhid, int Hdim, int Wdim)
{
    const long Mtot = (long)NB * Ntok;
    gemm(s, xn, Din, 0, 0, P.fc1_w, Din, 0, 0, P.fc1_b, nullptr, 0, 0,
         Hh, Dhid, 0, (int)Mtot, Dhid, Din, 1);
    dwconv(s, Hh, P.dw_w, P.dw_b, Cc, NB, Hdim, Wdim, Dhid);
    addk(s, Cc, Hh, Cc, Mtot * Dhid);
    ln(s, Cc, P.ln_w, P.ln_b, Cc, Mtot, Dhid);
    geluk(s, Cc, Mtot * Dhid);
    gemm(s, Cc, Dhid, 0, 0, P.fc2_w, Dhid, 0, 0, P.fc2_b, resid, Din, 0,
         out, Din, 0, (int)Mtot, Din, Dhid, 1);
}

static void run_dual(hipStream_t s, const DualP& P,
                     float* A, float* Bb, float* Nn, float* Hh, float* Cc,
                     float* X1, float* X2)
{
    const int Ntok = N_OUT, D = D_OUT, HEADS = 8, HD = D_OUT / 8;
    const long Mtot = (long)NB * Ntok;
    const long sTok = (long)Ntok * D;

    // n1 + efficient (linear) attention
    ln(s, A, P.n1_w, P.n1_b, Nn, Mtot, D);
    gemm(s, Nn, D, 0, 0, P.ea.k_w, D, 0, 0, P.ea.k_b, nullptr, 0, 0, X1, D, 0, (int)Mtot, D, D, 1);
    smax_mid(s, X1, X2, NB, Ntok, D);                                        // keys: softmax over N
    gemm(s, Nn, D, 0, 0, P.ea.q_w, D, 0, 0, P.ea.q_b, nullptr, 0, 0, X1, D, 0, (int)Mtot, D, D, 1);
    smax_last(s, X1, Cc, Mtot, D, nullptr, 1, 1);                            // qs: softmax over C
    gemm(s, Nn, D, 0, 0, P.ea.v_w, D, 0, 0, P.ea.v_b, nullptr, 0, 0, X1, D, 0, (int)Mtot, D, D, 1);
    // ctx[k,v] = sum_n keys[n,k] vs[n,v]
    gemm(s, X2, D, sTok, 1, X1, D, sTok, 1, nullptr, nullptr, 0, 0,
         Hh, D, (long)D * D, D, D, Ntok, NB);
    // att[n,v] = sum_k qs[n,k] ctx[k,v]
    gemm(s, Cc, D, sTok, 0, Hh, D, (long)D * D, 1, nullptr, nullptr, 0, 0,
         X2, D, sTok, Ntok, D, D, NB);
    gemm(s, X2, D, 0, 0, P.ea.r_w, D, 0, 0, P.ea.r_b, A, D, 0, Bb, D, 0, (int)Mtot, D, D, 1);

    // n2 + mixffn1
    ln(s, Bb, P.n2_w, P.n2_b, Nn, Mtot, D);
    run_mixffn(s, P.mlp1, Nn, Bb, A, Hh, Cc, Ntok, D, 4 * D, 64, 64);

    // n3 + channel attention
    ln(s, A, P.n3_w, P.n3_b, Nn, Mtot, D);
    gemm(s, Nn, D, 0, 0, P.ca.qkv_w, D, 0, 0, nullptr, nullptr, 0, 0,
         Hh, 3 * D, 0, (int)Mtot, 3 * D, D, 1);
    k_headprep<<<dim3(NB * HEADS * HD), 256, 0, s>>>(Hh, X1, X2, Cc, Ntok, D, HEADS);
    // G[d,e] = qn_d . kn_e  (per b,h): M=N=32, K=4096
    gemm(s, X1, Ntok, (long)HD * Ntok, 0, X2, Ntok, (long)HD * Ntok, 0,
         nullptr, nullptr, 0, 0, Hh, HD, (long)HD * HD, HD, HD, Ntok, NB * HEADS);
    smax_last(s, Hh, Hh, (long)NB * HEADS * HD, HD, P.ca.temp, HD, HEADS);   // *temp[h], softmax_e
    // out[d,n] = sum_e G[d,e] v[e,n]
    gemm(s, Hh, HD, (long)HD * HD, 0, Cc, Ntok, (long)HD * Ntok, 1,
         nullptr, nullptr, 0, 0, X1, Ntok, (long)HD * Ntok, HD, Ntok, HD, NB * HEADS);
    k_untranspose<<<dim3((unsigned)((Mtot * D + 255) / 256)), 256, 0, s>>>(X1, X2, NB, Ntok, D, HEADS);
    gemm(s, X2, D, 0, 0, P.ca.proj_w, D, 0, 0, P.ca.proj_b, A, D, 0, Bb, D, 0, (int)Mtot, D, D, 1);

    // n4 + mixffn2
    ln(s, Bb, P.n4_w, P.n4_b, Nn, Mtot, D);
    run_mixffn(s, P.mlp2, Nn, Bb, A, Hh, Cc, Ntok, D, 4 * D, 64, 64);
}

static void run_merging(hipStream_t s, const MabP& P,
                        const float* x, const float* x2, const float* x3,
                        float* A, float* Bb, float* Nn, float* Hh, float* Cc,
                        float* X1, float* X2)
{
    const long Min  = (long)NB * N_IN;    // 8192
    const long Mout = (long)NB * N_OUT;   // 32768
    const long sIn  = (long)N_IN * D_IN;
    const long sOut = (long)N_OUT * D_IN;

    // cross(x, x3)
    smax_last(s, x3, X1, Min, D_IN, nullptr, 1, 1);                // key1
    smax_mid(s, x3, X2, NB, N_IN, D_IN);                           // query1
    gemm(s, x, D_IN, sIn, 0, X1, D_IN, sIn, 0, nullptr, nullptr, 0, 0,
         Hh, N_IN, (long)N_IN * N_IN, N_IN, N_IN, D_IN, NB);       // ctx1
    gemm(s, Hh, N_IN, (long)N_IN * N_IN, 0, X2, D_IN, sIn, 1, nullptr, nullptr, 0, 0,
         Cc, D_IN, sIn, N_IN, D_IN, N_IN, NB);                     // att1
    gemm(s, Cc, D_IN, 0, 0, P.ca1_w, D_IN, 0, 0, P.ca1_b, nullptr, 0, 0,
         A, D_IN, 0, (int)Min, D_IN, D_IN, 1);                     // a1

    // cross(x, up)
    gemm(s, x2, D_OUT, 0, 0, P.lin_w, D_OUT, 0, 0, nullptr, nullptr, 0, 0,
         X1, D_IN, 0, (int)Mout, D_IN, D_OUT, 1);                  // up [B,4096,512]
    smax_last(s, X1, X2, Mout, D_IN, nullptr, 1, 1);               // key2
    smax_mid(s, X1, Cc, NB, N_OUT, D_IN);                          // query2
    gemm(s, x, D_IN, sIn, 0, X2, D_IN, sOut, 0, nullptr, nullptr, 0, 0,
         Hh, N_OUT, (long)N_IN * N_OUT, N_IN, N_OUT, D_IN, NB);    // ctx2 [1024,4096]
    gemm(s, Hh, N_OUT, (long)N_IN * N_OUT, 0, Cc, D_IN, sOut, 1, nullptr, nullptr, 0, 0,
         X1, D_IN, sIn, N_IN, D_IN, N_OUT, NB);                    // att2
    gemm(s, X1, D_IN, 0, 0, P.ca2_w, D_IN, 0, 0, P.ca2_b, nullptr, 0, 0,
         X2, D_IN, 0, (int)Min, D_IN, D_IN, 1);                    // a2

    k_merge3<<<dim3((unsigned)((Min * D_IN + 255) / 256)), 256, 0, s>>>(A, X2, x, P.w, Bb, Min * D_IN);
    ln(s, Bb, P.norm_w, P.norm_b, Nn, Min, D_IN);
    run_mixffn(s, P.ffn, Nn, Bb, A, Hh, Cc, N_IN, D_IN, 4 * D_IN, 32, 32);   // atten2 -> A

    gemm(s, A, D_IN, 0, 0, P.pe_lin_w, D_IN, 0, 0, nullptr, nullptr, 0, 0,
         X1, D_OUT, 0, (int)Min, D_OUT, D_IN, 1);                  // ex [B,1024,256]
    // ex2[o,c] = sum_n pe_conv[o,n] ex[n,c]
    gemm(s, P.pe_conv_w, N_IN, 0, 0, X1, D_OUT, (long)N_IN * D_OUT, 1,
         nullptr, nullptr, 0, 0, Bb, D_OUT, (long)N_OUT * D_OUT,
         N_OUT, D_OUT, N_IN, NB);
    ln(s, Bb, P.pe_norm_w, P.pe_norm_b, Bb, Mout, D_OUT);
    addk(s, x2, Bb, A, Mout * D_OUT);                              // A = x2 + ex
}

// ---------------------------------------------------------------------------
extern "C" void kernel_launch(void* const* d_in, const int* in_sizes, int n_in,
                              void* d_out, int out_size, void* d_ws, size_t ws_size,
                              hipStream_t stream)
{
    (void)in_sizes; (void)n_in; (void)out_size; (void)ws_size;
    Rd rd{d_in, 0};
    const float* x  = rd.next();
    const float* x2 = rd.next();
    const float* x3 = rd.next();
    DualP d1 = rdDual(rd);
    DualP d2 = rdDual(rd);
    const float* fn_b = rd.next();
    const float* fn_w = rd.next();
    MabP mab = rdMab(rd);

    float* W = (float*)d_ws;
    size_t o = 0;
    float* A  = W + o; o += 8UL * 4096 * 256;    // main activation (ping)
    float* Bb = W + o; o += 8UL * 4096 * 256;    // pong
    float* Nn = W + o; o += 8UL * 4096 * 512;    // normed (needs 512 width in merging)
    float* Hh = W + o; o += 8UL * 4096 * 1024;   // hidden / big scratch
    float* Cc = W + o; o += 8UL * 4096 * 1024;   // hidden / big scratch
    float* X1 = W + o; o += 8UL * 4096 * 512;
    float* X2 = W + o; o += 8UL * 4096 * 512;

    run_merging(stream, mab, x, x2, x3, A, Bb, Nn, Hh, Cc, X1, X2);
    run_dual(stream, d1, A, Bb, Nn, Hh, Cc, X1, X2);
    run_dual(stream, d2, A, Bb, Nn, Hh, Cc, X1, X2);
    ln(stream, A, fn_w, fn_b, (float*)d_out, 8L * 4096, 256);
}
